// MultiHeadAttention_3453153706467
// MI455X (gfx1250) — compile-verified, matrix operand order checked by reading, not run on hardware
//
#include <hip/hip_runtime.h>
#include <hip/hip_bf16.h>
#include <stdint.h>

// Problem constants (match the JAX reference)
#define HEADS  16
#define DMODEL 1024
#define DK     64
#define BATCH  4
#define SEQL   1024
#define MROWS  (BATCH * SEQL)   // 4096
#define NEG_BIG (-1000000000.0f)

typedef __bf16    bf16;
typedef uint32_t  u32;
typedef __attribute__((ext_vector_type(16))) bf16  v16bf;
typedef __attribute__((ext_vector_type(8)))  float v8f;
typedef __attribute__((ext_vector_type(4)))  u32   v4u;

union FragU { v16bf bf; v4u u[2]; };

// Load one 16-bit WMMA operand fragment: two contiguous 8-element (16B) chunks.
// Per the CDNA5 ISA 16-bit A-matrix layout, lane L holds K elements
// {half*8 .. half*8+7} and {16+half*8 .. 16+half*8+7} (half = L>=16).
// The caller passes pointers already offset to those two chunks.
__device__ inline v16bf load_frag(const bf16* p0, const bf16* p1) {
  FragU f;
  f.u[0] = *(const v4u*)p0;
  f.u[1] = *(const v4u*)p1;
  return f.bf;
}

__device__ inline v8f wmma_bf16(v16bf a, v16bf b, v8f c) {
  // D = A(16x32 bf16) x B(32x16 bf16) + C(16x16 f32)
  return __builtin_amdgcn_wmma_f32_16x16x32_bf16(
      false, a, false, b, (short)0, c, false, false);
}

// float -> bf16, round to nearest even (bit manipulation: no __bf16 arith needed)
__device__ inline bf16 f2b(float f) {
  union { float f; u32 u; } x; x.f = f;
  u32 r = x.u + 0x7FFFu + ((x.u >> 16) & 1u);
  union { unsigned short s; bf16 b; } y; y.s = (unsigned short)(r >> 16);
  return y.b;
}

__device__ inline float redmax16(float v) {
  v = fmaxf(v, __shfl_xor(v, 1, 16));
  v = fmaxf(v, __shfl_xor(v, 2, 16));
  v = fmaxf(v, __shfl_xor(v, 4, 16));
  v = fmaxf(v, __shfl_xor(v, 8, 16));
  return v;
}
__device__ inline float redsum16(float v) {
  v += __shfl_xor(v, 1, 16);
  v += __shfl_xor(v, 2, 16);
  v += __shfl_xor(v, 4, 16);
  v += __shfl_xor(v, 8, 16);
  return v;
}

// ---------------------------------------------------------------------------
// float32 -> bf16 conversion
// ---------------------------------------------------------------------------
__global__ void mha_f32_to_bf16(const float* __restrict__ x,
                                bf16* __restrict__ y, int n) {
  int i = blockIdx.x * blockDim.x + threadIdx.x;
  if (i < n) y[i] = f2b(x[i]);
}

// ---------------------------------------------------------------------------
// C[4096,1024] = A[4096,1024](bf16) @ W[1024,1024](bf16)^T
// MODE 0: store bf16 row-major
// MODE 1: scatter-store bf16 into per-head V^T layout [b,h][64][1024]
//         (Vh head slab is the contiguous reshape of P[b, 64h:64h+64, :])
// MODE 2: store f32 = C + resid (final output projection + residual)
// Wave tile 32x64 (2x4 WMMA accumulators); workgroup (8 waves) tile 64x256.
// Because C = A*W^T, both WMMA operands use the identical contiguous
// per-lane load pattern from row-major A and W.
// ---------------------------------------------------------------------------
template <int MODE>
__global__ __launch_bounds__(256)
void mha_gemm_bf16(const bf16* __restrict__ A, const bf16* __restrict__ W,
                   bf16* __restrict__ outB, float* __restrict__ outF,
                   const float* __restrict__ resid) {
  const int lane = threadIdx.x & 31;
  const int wave = threadIdx.x >> 5;
  const int half = lane >> 4;
  const int r    = lane & 15;
  const int wm   = wave >> 2;  // 0..1
  const int wn   = wave & 3;   // 0..3
  const int M0 = blockIdx.y * 64 + wm * 32;
  const int N0 = blockIdx.x * 256 + wn * 64;

  v8f acc[2][4];
#pragma unroll
  for (int i = 0; i < 2; ++i)
#pragma unroll
    for (int j = 0; j < 4; ++j) acc[i][j] = (v8f){0, 0, 0, 0, 0, 0, 0, 0};

  for (int k0 = 0; k0 < DMODEL; k0 += 32) {
    const int c0 = k0 + half * 8;
    const int c1 = k0 + 16 + half * 8;
    v16bf af[2], bf_[4];
#pragma unroll
    for (int ms = 0; ms < 2; ++ms) {
      const bf16* arow = A + (size_t)(M0 + ms * 16 + r) * DMODEL;
      af[ms] = load_frag(arow + c0, arow + c1);
    }
#pragma unroll
    for (int ns = 0; ns < 4; ++ns) {
      const bf16* wrow = W + (size_t)(N0 + ns * 16 + r) * DMODEL;
      bf_[ns] = load_frag(wrow + c0, wrow + c1);
    }
#pragma unroll
    for (int ms = 0; ms < 2; ++ms)
#pragma unroll
      for (int ns = 0; ns < 4; ++ns)
        acc[ms][ns] = wmma_bf16(af[ms], bf_[ns], acc[ms][ns]);
  }

  // C/D layout: lanes 0-15 hold rows j, lanes 16-31 rows j+8; col = lane&15.
#pragma unroll
  for (int ms = 0; ms < 2; ++ms) {
#pragma unroll
    for (int ns = 0; ns < 4; ++ns) {
#pragma unroll
      for (int j = 0; j < 8; ++j) {
        const int i   = M0 + ms * 16 + j + 8 * half;  // global row 0..4095
        const int col = N0 + ns * 16 + r;             // 0..1023
        const float vv = acc[ms][ns][j];
        if (MODE == 0) {
          outB[(size_t)i * DMODEL + col] = f2b(vv);
        } else if (MODE == 1) {
          const int b = i >> 10, s = i & 1023;
          const int h = s >> 6;
          const int d = col & 63;
          const int t = ((s & 63) << 4) + (col >> 6);
          outB[(((size_t)(b * HEADS + h) * DK) + d) * SEQL + t] = f2b(vv);
        } else {
          const size_t idx = (size_t)i * DMODEL + col;
          outF[idx] = vv + resid[idx];
        }
      }
    }
  }
}

// ---------------------------------------------------------------------------
// Attention: per (b,h) the head matrices are contiguous [1024,64] slabs.
// 8 waves / WG; each wave owns 16 q-rows. Exact two-pass softmax:
//   pass 1: stream Q*K^T tiles, track per-row max (mask raw==0 -> -1e9, /8)
//   pass 2: recompute tiles, p = exp(s - m), accumulate row sum and O = P*V
// P tiles bounce through LDS (C-layout -> A-fragment layout conversion).
// V is consumed from the precomputed V^T layout so b-operands are contiguous.
// ---------------------------------------------------------------------------
__global__ __launch_bounds__(256)
void mha_attn(const bf16* __restrict__ Pq, const bf16* __restrict__ Pk,
              const bf16* __restrict__ Vt, bf16* __restrict__ Pc) {
  __shared__ bf16 Plds[8][16][40];  // stride 40 -> conflict-free b128 reads

  const int lane = threadIdx.x & 31;
  const int wave = threadIdx.x >> 5;
  const int half = lane >> 4;
  const int r    = lane & 15;

  const int qblk = blockIdx.x & 7;
  const int bh   = blockIdx.x >> 3;
  const int b    = bh >> 4;
  const int h    = bh & 15;

  const int q0 = qblk * 128 + wave * 16;

  const bf16* Qh  = Pq + (size_t)bh * (SEQL * DK);  // [1024][64]
  const bf16* Kh  = Pk + (size_t)bh * (SEQL * DK);  // [1024][64]
  const bf16* Vth = Vt + (size_t)bh * (DK * SEQL);  // [64][1024]

  // Q fragments for this wave's 16 rows (K-dim = d, chunks 0..31 and 32..63)
  v16bf aq[2];
  {
    const bf16* qrow = Qh + (size_t)(q0 + r) * DK;
    aq[0] = load_frag(qrow + half * 8, qrow + 16 + half * 8);
    aq[1] = load_frag(qrow + 32 + half * 8, qrow + 48 + half * 8);
  }

  const v8f vzero = (v8f){0, 0, 0, 0, 0, 0, 0, 0};

  // ---- pass 1: row maxima ----
  float m[8];
#pragma unroll
  for (int j = 0; j < 8; ++j) m[j] = -1e30f;

  for (int kb = 0; kb < SEQL; kb += 16) {
    const bf16* krow = Kh + (size_t)(kb + r) * DK;
    v16bf bk0 = load_frag(krow + half * 8, krow + 16 + half * 8);
    v16bf bk1 = load_frag(krow + 32 + half * 8, krow + 48 + half * 8);
    v8f s = vzero;
    s = wmma_bf16(aq[0], bk0, s);
    s = wmma_bf16(aq[1], bk1, s);
#pragma unroll
    for (int j = 0; j < 8; ++j) {
      float sv = s[j];
      sv = (sv == 0.0f) ? NEG_BIG : sv * 0.125f;  // mask, then 1/sqrt(64)
      m[j] = fmaxf(m[j], sv);
    }
  }
#pragma unroll
  for (int j = 0; j < 8; ++j) m[j] = redmax16(m[j]);

  // ---- pass 2: sum + O accumulation ----
  float l[8];
#pragma unroll
  for (int j = 0; j < 8; ++j) l[j] = 0.0f;
  v8f o[4];
#pragma unroll
  for (int nb = 0; nb < 4; ++nb) o[nb] = vzero;

  for (int kb = 0; kb < SEQL; kb += 32) {
    v8f s[2];
#pragma unroll
    for (int t = 0; t < 2; ++t) {
      const bf16* krow = Kh + (size_t)(kb + t * 16 + r) * DK;
      v16bf bk0 = load_frag(krow + half * 8, krow + 16 + half * 8);
      v16bf bk1 = load_frag(krow + 32 + half * 8, krow + 48 + half * 8);
      s[t] = vzero;
      s[t] = wmma_bf16(aq[0], bk0, s[t]);
      s[t] = wmma_bf16(aq[1], bk1, s[t]);
    }

    __syncthreads();  // protect previous iteration's LDS reads
#pragma unroll
    for (int t = 0; t < 2; ++t) {
#pragma unroll
      for (int j = 0; j < 8; ++j) {
        float sv = s[t][j];
        sv = (sv == 0.0f) ? NEG_BIG : sv * 0.125f;
        float p = __expf(sv - m[j]);
        l[j] += p;
        Plds[wave][j + 8 * half][t * 16 + r] = f2b(p);
      }
    }
    __syncthreads();  // publish P tile (C-layout -> row-major in LDS)

    // Reload P as A-fragment (16 rows x 32 k), contiguous ds_load_b128.
    const bf16* prow = &Plds[wave][r][0];
    v16bf ap = load_frag(prow + half * 8, prow + 16 + half * 8);

#pragma unroll
    for (int nb = 0; nb < 4; ++nb) {
      const bf16* vrow = Vth + (size_t)(nb * 16 + r) * SEQL + kb;
      v16bf bv = load_frag(vrow + half * 8, vrow + 16 + half * 8);
      o[nb] = wmma_bf16(ap, bv, o[nb]);
    }
  }

#pragma unroll
  for (int j = 0; j < 8; ++j) l[j] = redsum16(l[j]);

  // Store O into concat layout: Pc[b][t][h*64 + d]
#pragma unroll
  for (int nb = 0; nb < 4; ++nb) {
#pragma unroll
    for (int j = 0; j < 8; ++j) {
      const int row = q0 + j + 8 * half;
      const int col = h * DK + nb * 16 + r;
      Pc[((size_t)b * SEQL + row) * DMODEL + col] = f2b(o[nb][j] / l[j]);
    }
  }
}

// ---------------------------------------------------------------------------
// Host-side launch. Workspace layout (needs 64 MB):
//  [ 0MB) qb  [ 8MB) kb  [16MB) vb            (bf16 activations, 8MB each)
//  [24MB) Wqb [26MB) Wkb [28MB) Wvb [30MB) Wob (bf16 weights, 2MB each)
//  [32MB) Pq  [40MB) Pk  [48MB) Vt  [56MB) Pc  (bf16 intermediates, 8MB each)
// ---------------------------------------------------------------------------
extern "C" void kernel_launch(void* const* d_in, const int* in_sizes, int n_in,
                              void* d_out, int out_size, void* d_ws,
                              size_t ws_size, hipStream_t stream) {
  (void)in_sizes; (void)n_in; (void)out_size; (void)ws_size;
  const float* q  = (const float*)d_in[0];
  const float* k  = (const float*)d_in[1];
  const float* v  = (const float*)d_in[2];
  const float* Wq = (const float*)d_in[3];
  const float* Wk = (const float*)d_in[4];
  const float* Wv = (const float*)d_in[5];
  const float* Wo = (const float*)d_in[6];
  float* out = (float*)d_out;

  char* ws = (char*)d_ws;
  const size_t MB = (size_t)1 << 20;
  bf16* qb  = (bf16*)(ws + 0 * MB);
  bf16* kb  = (bf16*)(ws + 8 * MB);
  bf16* vb  = (bf16*)(ws + 16 * MB);
  bf16* Wqb = (bf16*)(ws + 24 * MB);
  bf16* Wkb = (bf16*)(ws + 26 * MB);
  bf16* Wvb = (bf16*)(ws + 28 * MB);
  bf16* Wob = (bf16*)(ws + 30 * MB);
  bf16* Pq  = (bf16*)(ws + 32 * MB);
  bf16* Pk  = (bf16*)(ws + 40 * MB);
  bf16* Vt  = (bf16*)(ws + 48 * MB);
  bf16* Pc  = (bf16*)(ws + 56 * MB);

  const int nAct = MROWS * DMODEL;   // 4,194,304
  const int nW   = DMODEL * DMODEL;  // 1,048,576
  mha_f32_to_bf16<<<nAct / 256, 256, 0, stream>>>(q, qb, nAct);
  mha_f32_to_bf16<<<nAct / 256, 256, 0, stream>>>(k, kb, nAct);
  mha_f32_to_bf16<<<nAct / 256, 256, 0, stream>>>(v, vb, nAct);
  mha_f32_to_bf16<<<nW / 256, 256, 0, stream>>>(Wq, Wqb, nW);
  mha_f32_to_bf16<<<nW / 256, 256, 0, stream>>>(Wk, Wkb, nW);
  mha_f32_to_bf16<<<nW / 256, 256, 0, stream>>>(Wv, Wvb, nW);
  mha_f32_to_bf16<<<nW / 256, 256, 0, stream>>>(Wo, Wob, nW);

  dim3 ggrid(DMODEL / 256, MROWS / 64);  // (4, 64)
  mha_gemm_bf16<0><<<ggrid, 256, 0, stream>>>(qb, Wqb, Pq, nullptr, nullptr);
  mha_gemm_bf16<0><<<ggrid, 256, 0, stream>>>(kb, Wkb, Pk, nullptr, nullptr);
  mha_gemm_bf16<1><<<ggrid, 256, 0, stream>>>(vb, Wvb, Vt, nullptr, nullptr);

  mha_attn<<<BATCH * HEADS * (SEQL / 128), 256, 0, stream>>>(Pq, Pk, Vt, Pc);

  mha_gemm_bf16<2><<<ggrid, 256, 0, stream>>>(Pc, Wob, nullptr, out, q);
}
